// TensorNet_78237124264574
// MI455X (gfx1250) — matrix operand order repaired and checked by reading, not compiled, not run on hardware
//
#include <hip/hip_runtime.h>
#include <hip/hip_bf16.h>

typedef __attribute__((ext_vector_type(16))) _Float16 v16h;
typedef __attribute__((ext_vector_type(8)))  _Float16 v8h;
typedef __attribute__((ext_vector_type(8)))  float    v8f;

// Problem constants
#define C_NIN     128
#define C_NHID    256
#define C_NDEC    512
#define C_NOUT    10
#define C_B       32
#define C_N       4096
#define C_ROWS    (C_B * C_N)        // 131072

// Encoder tiling
#define TM        64                 // rows per block
#define NTHREADS  256                // 8 waves
#define NWAVES    8
#define XP        (C_NIN + 8)        // padded LDS stride (halves) for x tile
#define AP        (C_NHID + 8)       // padded LDS stride (halves) for activations

// Packed-weight geometry: tiles of 32(K) x 16(N), 512 halves each, lane-major
#define PW1_OFF   0                  // W1: (128/32)*(256/16) = 64 tiles
#define PW2_OFF   32768              // W2: 8*16 = 128 tiles
#define PW3_OFF   98304              // W3: 128 tiles
#define PW_TOTAL  163840             // halves

// ---------------------------------------------------------------------------
// Weight packing: f32 row-major [K][256] -> f16 B-fragment tiles.
// Tile (kt,nt): element (L,j) = W[kt*32 + (L/16)*8 + (j&7) + (j/8)*16][nt*16 + (L&15)]
// so each lane's 16-half B fragment is one contiguous 32B chunk.
// ---------------------------------------------------------------------------
__global__ __launch_bounds__(256) void pack_weights(
    const float* __restrict__ W1, const float* __restrict__ W2,
    const float* __restrict__ W3, _Float16* __restrict__ out)
{
    int e = blockIdx.x * 256 + threadIdx.x;
    if (e >= PW_TOTAL) return;
    const float* W; int base;
    if (e < PW2_OFF)      { W = W1; base = PW1_OFF; }
    else if (e < PW3_OFF) { W = W2; base = PW2_OFF; }
    else                  { W = W3; base = PW3_OFF; }
    int local = e - base;
    int t  = local >> 9;         // tile index
    int r  = local & 511;
    int L  = r >> 4;             // lane 0..31
    int j  = r & 15;             // K within lane-half
    int kt = t >> 4;             // 16 n-tiles per K-slab
    int nt = t & 15;
    int k  = kt * 32 + ((L >> 4) << 3) + (j & 7) + ((j >> 3) << 4);
    int n  = nt * 16 + (L & 15);
    out[e] = (_Float16)W[k * C_NHID + n];
}

// ---------------------------------------------------------------------------
// A-fragment gather from LDS (16-bit A 16x32 layout):
// lane L, row m0+(L&15); halves [0..7] = K kbase+(L/16)*8+{0..7},
//                        halves [8..15] = K kbase+16+(L/16)*8+{0..7}
// ---------------------------------------------------------------------------
__device__ __forceinline__ v16h load_afrag(const _Float16* act, int stride,
                                           int m0, int kbase, int lane)
{
    const _Float16* p = act + (m0 + (lane & 15)) * stride
                            + kbase + ((lane >> 4) << 3);
    v8h lo = *(const v8h*)p;
    v8h hi = *(const v8h*)(p + 16);
    return __builtin_shufflevector(lo, hi, 0,1,2,3,4,5,6,7,8,9,10,11,12,13,14,15);
}

// ---------------------------------------------------------------------------
// One encoder layer for a 64-row tile.
// Wave w owns row-half rh = w&1 (two 16-row groups) and n-tiles
// [ (w>>1)*4 , (w>>1)*4+4 ). A fragments for both row groups and the whole
// K dimension are register-resident; each packed B fragment is loaded once
// from L2 and feeds TWO wmma (one per row group) -> 32 FLOP per L2 byte.
// STORE=true  -> relu+f16 store to outAct
// STORE=false -> row-reduce into ps (layer 3)
// ---------------------------------------------------------------------------
template<int KDIM, bool STORE>
__device__ __forceinline__ void layer(
    const _Float16* __restrict__ inAct, int inStride,
    _Float16* __restrict__ outAct,
    const _Float16* __restrict__ pW,
    const float* __restrict__ bvec,
    float* __restrict__ ps,
    int lane, int wid)
{
    const int rh = wid & 1;          // row half: rows rh*32 .. rh*32+31
    const int nq = wid >> 1;         // n-quarter: 4 n-tiles

    v16h afrag[2][KDIM / 32];
#pragma unroll
    for (int g = 0; g < 2; ++g)
#pragma unroll
        for (int kt = 0; kt < KDIM / 32; ++kt)
            afrag[g][kt] = load_afrag(inAct, inStride, (rh * 2 + g) * 16,
                                      kt * 32, lane);

#pragma unroll 2
    for (int ntl = 0; ntl < 4; ++ntl) {
        const int nt = nq * 4 + ntl;
        v8f acc0 = {}, acc1 = {};
#pragma unroll
        for (int kt = 0; kt < KDIM / 32; ++kt) {
            v16h b = *(const v16h*)(pW + ((size_t)(kt * 16 + nt) << 9)
                                       + (lane << 4));
            acc0 = __builtin_amdgcn_wmma_f32_16x16x32_f16(
                false, afrag[0][kt], false, b, (short)0, acc0, false, false);
            acc1 = __builtin_amdgcn_wmma_f32_16x16x32_f16(
                false, afrag[1][kt], false, b, (short)0, acc1, false, false);
        }
        const float bias = bvec[nt * 16 + (lane & 15)];
        const int col = nt * 16 + (lane & 15);
        if (STORE) {
            const int rb0 = (rh * 2 + 0) * 16 + ((lane >> 4) << 3);
            const int rb1 = (rh * 2 + 1) * 16 + ((lane >> 4) << 3);
#pragma unroll
            for (int i = 0; i < 8; ++i) {
                float v0 = acc0[i] + bias; v0 = v0 > 0.f ? v0 : 0.f;
                float v1 = acc1[i] + bias; v1 = v1 > 0.f ? v1 : 0.f;
                outAct[(rb0 + i) * AP + col] = (_Float16)v0;
                outAct[(rb1 + i) * AP + col] = (_Float16)v1;
            }
        } else {
            float s0 = 0.f, s1 = 0.f;
#pragma unroll
            for (int i = 0; i < 8; ++i) {
                float v0 = acc0[i] + bias; v0 = v0 > 0.f ? v0 : 0.f;
                float v1 = acc1[i] + bias; v1 = v1 > 0.f ? v1 : 0.f;
                s0 += v0; s1 += v1;
            }
            s0 += __shfl_xor(s0, 16, 32);    // combine the two lane halves
            s1 += __shfl_xor(s1, 16, 32);
            if (lane < 16) {
                ps[(rh * 2 + 0) * C_NHID + col] = s0;
                ps[(rh * 2 + 1) * C_NHID + col] = s1;
            }
        }
    }
}

// ---------------------------------------------------------------------------
// Fused 3-layer encoder + per-block row-sum. One block = 64 rows.
// ---------------------------------------------------------------------------
__global__ __launch_bounds__(NTHREADS) void encoder(
    const float* __restrict__ x,
    const _Float16* __restrict__ pW1, const _Float16* __restrict__ pW2,
    const _Float16* __restrict__ pW3,
    const float* __restrict__ b1, const float* __restrict__ b2,
    const float* __restrict__ b3,
    float* __restrict__ partials)
{
    extern __shared__ char smem[];
    _Float16* xh  = (_Float16*)smem;                                    // 64*136 h
    _Float16* a1  = (_Float16*)(smem + TM * XP * 2);                    // 64*264 h
    _Float16* a2  = (_Float16*)(smem + TM * XP * 2 + TM * AP * 2);      // 64*264 h
    float*    ps  = (float*)   (smem + TM * XP * 2 + 2 * TM * AP * 2);  // 4*256 f

    const int tid  = threadIdx.x;
    const int lane = tid & 31;
    const int wid  = tid >> 5;
    const int r0   = blockIdx.x * TM;

    // Stage x tile f32 -> f16 (coalesced float4 loads)
    for (int i = tid; i < TM * (C_NIN / 4); i += NTHREADS) {
        int row = i / (C_NIN / 4);
        int c4  = (i % (C_NIN / 4)) * 4;
        float4 v = *(const float4*)(x + (size_t)(r0 + row) * C_NIN + c4);
        _Float16* dst = xh + row * XP + c4;
        dst[0] = (_Float16)v.x; dst[1] = (_Float16)v.y;
        dst[2] = (_Float16)v.z; dst[3] = (_Float16)v.w;
    }
    __syncthreads();

    layer<C_NIN,  true >(xh, XP, a1, pW1, b1, ps, lane, wid);      // L1
    __syncthreads();
    layer<C_NHID, true >(a1, AP, a2, pW2, b2, ps, lane, wid);      // L2
    __syncthreads();
    layer<C_NHID, false>(a2, AP, nullptr, pW3, b3, ps, lane, wid); // L3 + reduce
    __syncthreads();

    // Per-block partial sum per feature (fixed order -> deterministic)
    float tot = ps[0 * C_NHID + tid] + ps[1 * C_NHID + tid]
              + ps[2 * C_NHID + tid] + ps[3 * C_NHID + tid];
    partials[(size_t)blockIdx.x * C_NHID + tid] = tot;
}

// ---------------------------------------------------------------------------
// Deterministic reduction: 64 blocks per batch -> feature sums
// ---------------------------------------------------------------------------
__global__ __launch_bounds__(256) void reduce_partials(
    const float* __restrict__ partials, float* __restrict__ accum)
{
    int b = blockIdx.x, f = threadIdx.x;
    const float* p = partials + ((size_t)b * (C_N / TM)) * C_NHID + f;
    float s = 0.f;
#pragma unroll 8
    for (int i = 0; i < C_N / TM; ++i) s += p[(size_t)i * C_NHID];
    accum[b * C_NHID + f] = s;
}

// ---------------------------------------------------------------------------
// Decoder (32 rows -> plain f32 FMA; GEMM hardware wasted at this size)
// ---------------------------------------------------------------------------
__global__ __launch_bounds__(256) void decode1(
    const float* __restrict__ accum, const float* __restrict__ D1,
    const float* __restrict__ c1, float* __restrict__ d1)
{
    __shared__ float pbuf[C_NHID];
    int r = blockIdx.x, tid = threadIdx.x;
    float m = accum[r * C_NHID + tid] * (1.0f / (float)C_N);
    pbuf[tid] = m * m;                       // relu(m*m) == m*m
    __syncthreads();
    float s0 = 0.f, s1 = 0.f;
    for (int k = 0; k < C_NHID; ++k) {
        float pk = pbuf[k];
        s0 = fmaf(pk, D1[k * C_NDEC + tid], s0);
        s1 = fmaf(pk, D1[k * C_NDEC + tid + 256], s1);
    }
    s0 += c1[tid]; s1 += c1[tid + 256];
    d1[r * C_NDEC + tid]       = s0 > 0.f ? s0 : 0.f;
    d1[r * C_NDEC + tid + 256] = s1 > 0.f ? s1 : 0.f;
}

__global__ __launch_bounds__(256) void decode2(
    const float* __restrict__ d1, const float* __restrict__ D2,
    const float* __restrict__ c2, float* __restrict__ d2)
{
    __shared__ float abuf[C_NDEC];
    int r = blockIdx.x, tid = threadIdx.x;
    abuf[tid]       = d1[r * C_NDEC + tid];
    abuf[tid + 256] = d1[r * C_NDEC + tid + 256];
    __syncthreads();
    float s0 = 0.f, s1 = 0.f;
    for (int k = 0; k < C_NDEC; ++k) {
        float ak = abuf[k];
        s0 = fmaf(ak, D2[k * C_NDEC + tid], s0);
        s1 = fmaf(ak, D2[k * C_NDEC + tid + 256], s1);
    }
    s0 += c2[tid]; s1 += c2[tid + 256];
    d2[r * C_NDEC + tid]       = s0 > 0.f ? s0 : 0.f;
    d2[r * C_NDEC + tid + 256] = s1 > 0.f ? s1 : 0.f;
}

__global__ __launch_bounds__(64) void decode3(
    const float* __restrict__ d2, const float* __restrict__ D3,
    const float* __restrict__ c3, float* __restrict__ out)
{
    int t = blockIdx.x * 64 + threadIdx.x;
    if (t >= C_B * C_NOUT) return;
    int r = t / C_NOUT, n = t % C_NOUT;
    float s = c3[n];
    for (int k = 0; k < C_NDEC; ++k)
        s = fmaf(d2[r * C_NDEC + k], D3[k * C_NOUT + n], s);
    out[t] = s;
}

// ---------------------------------------------------------------------------
extern "C" void kernel_launch(void* const* d_in, const int* in_sizes, int n_in,
                              void* d_out, int out_size, void* d_ws, size_t ws_size,
                              hipStream_t stream)
{
    const float* x  = (const float*)d_in[0];
    const float* W1 = (const float*)d_in[1];
    const float* b1 = (const float*)d_in[2];
    const float* W2 = (const float*)d_in[3];
    const float* b2 = (const float*)d_in[4];
    const float* W3 = (const float*)d_in[5];
    const float* b3 = (const float*)d_in[6];
    const float* D1 = (const float*)d_in[7];
    const float* c1 = (const float*)d_in[8];
    const float* D2 = (const float*)d_in[9];
    const float* c2 = (const float*)d_in[10];
    const float* D3 = (const float*)d_in[11];
    const float* c3 = (const float*)d_in[12];

    char* ws = (char*)d_ws;
    _Float16* packW   = (_Float16*)ws;                          // 327,680 B
    float*   partials = (float*)(ws + 327680);                  // 2048*256*4 = 2 MB
    float*   accum    = (float*)(ws + 327680 + 2097152);        // 32 KB
    float*   d1buf    = (float*)(ws + 327680 + 2097152 + 32768);      // 64 KB
    float*   d2buf    = (float*)(ws + 327680 + 2097152 + 32768 + 65536);

    pack_weights<<<(PW_TOTAL + 255) / 256, 256, 0, stream>>>(W1, W2, W3, packW);

    const size_t smem = (size_t)TM * XP * 2 + 2 * (size_t)TM * AP * 2
                      + 4 * C_NHID * sizeof(float);             // 89,088 B
    encoder<<<C_ROWS / TM, NTHREADS, smem, stream>>>(
        x, packW + PW1_OFF, packW + PW2_OFF, packW + PW3_OFF,
        b1, b2, b3, partials);

    reduce_partials<<<C_B, C_NHID, 0, stream>>>(partials, accum);
    decode1<<<C_B, 256, 0, stream>>>(accum, D1, c1, d1buf);
    decode2<<<C_B, 256, 0, stream>>>(d1buf, D2, c2, d2buf);
    decode3<<<(C_B * C_NOUT + 63) / 64, 64, 0, stream>>>(d2buf, D3, c3, (float*)d_out);
}
